// Sae_12661563589103
// MI455X (gfx1250) — matrix operand order repaired and checked by reading, not compiled, not run on hardware
//
#include <hip/hip_runtime.h>
#include <stdint.h>

#define D_IN   768
#define D_LAT  12288
#define NROWS  16384
#define KTOP   20
#define NKSTEP (D_IN / 32)     // 24 K-steps of 32 for bf16 WMMA
#define NTILE  (D_LAT / 16)    // 768 column tiles of 16
#define NHALF  (NTILE / 2)     // 384 tiles per N-half
#define MWAVE  32              // rows per strip (two 16-row WMMA tiles)
#define WPB    4               // waves per block: 2 strips x 2 N-halves

typedef __attribute__((ext_vector_type(16))) __bf16 v16bf;
typedef __attribute__((ext_vector_type(8)))  float  v8f;

union BF16Vec {
    v16bf v;
    uint4 q[2];
};

// ---------------------------------------------------------------------------
// Prep kernel 1: fp32 -> bf16 (round-to-nearest-even), 2 elements per thread
// ---------------------------------------------------------------------------
__global__ void cvt_bf16_pk(const float* __restrict__ in,
                            uint32_t* __restrict__ out, int n2) {
    int i = blockIdx.x * blockDim.x + threadIdx.x;
    int stride = gridDim.x * blockDim.x;
    for (; i < n2; i += stride) {
        uint32_t ua = __float_as_uint(in[2 * i]);
        uint32_t ub = __float_as_uint(in[2 * i + 1]);
        ua = (ua + 0x7FFFu + ((ua >> 16) & 1u)) >> 16;
        ub = (ub + 0x7FFFu + ((ub >> 16) & 1u)) >> 16;
        out[i] = ua | (ub << 16);
    }
}

// ---------------------------------------------------------------------------
// Prep kernel 2: W_dec [768][12288] -> Wt [12288][768] (LDS-tiled transpose)
// grid = (D_LAT/32, D_IN/32), block = (32, 8)
// ---------------------------------------------------------------------------
__global__ void transpose_f32(const float* __restrict__ W,
                              float* __restrict__ Wt) {
    __shared__ float tile[32][33];
    const int bx = blockIdx.x;   // along D_LAT
    const int by = blockIdx.y;   // along D_IN
    const int tx = threadIdx.x;  // 0..31
    const int ty = threadIdx.y;  // 0..7
#pragma unroll
    for (int i = 0; i < 4; ++i)
        tile[ty + i * 8][tx] =
            W[(size_t)(by * 32 + ty + i * 8) * D_LAT + bx * 32 + tx];
    __syncthreads();
#pragma unroll
    for (int i = 0; i < 4; ++i)
        Wt[(size_t)(bx * 32 + ty + i * 8) * D_IN + by * 32 + tx] =
            tile[tx][ty + i * 8];
}

// ---------------------------------------------------------------------------
// Fused encoder GEMM (bf16 WMMA, fp32 acc, M=32) + streaming top-20 + merge +
// sparse decoder.  Each 32-row strip is covered by TWO waves, one per half of
// the 768 column tiles (1024 waves total -> 2 waves/SIMD of latency cover).
// A block holds 2 strips x 2 halves; the per-tile barrier keeps the two
// same-half waves in lockstep so their B streams stay WGP$-resident.
// ---------------------------------------------------------------------------
__global__ __launch_bounds__(WPB * 32, 1) void sae_fused(
    const uint16_t* __restrict__ xB,     // [16384][768] bf16
    const uint16_t* __restrict__ WencB,  // [12288][768] bf16
    const float*    __restrict__ b_enc,  // [12288]
    const float*    __restrict__ Wt,     // [12288][768] = W_dec^T, fp32
    const float*    __restrict__ b_dec,  // [768]
    float*          __restrict__ out)    // [16384][768]
{
    const int lane = threadIdx.x & 31;
    const int wave = threadIdx.x >> 5;        // 0..3
    const int sidx = wave & 1;                // strip within block
    const int half = wave >> 1;               // N-half (0: tiles 0..383)
    const int strip = blockIdx.x * 2 + sidx;  // 0..511
    const int row0 = strip * MWAVE;

    __shared__ float s_tile[WPB][MWAVE][16];
    __shared__ float s_topv[2][2][MWAVE][KTOP];   // [strip][half][row][k]
    __shared__ int   s_topi[2][2][MWAVE][KTOP];
    __shared__ float s_mv[2][MWAVE][KTOP];        // merged lists
    __shared__ int   s_mi[2][MWAVE][KTOP];

    // Every lane owns one row (row = lane) for the streaming top-20.
#pragma unroll
    for (int s = 0; s < KTOP; ++s) {
        s_topv[sidx][half][lane][s] = -__builtin_inff();
        s_topi[sidx][half][lane][s] = 0;
    }
    float thr = -__builtin_inff();   // list[KTOP-1] cached in a VGPR

    // ---- Load two 16x768 A-tiles (rows row0..row0+31) into registers in
    //      native WMMA 16-bit A layout:
    //      lanes 0-15 : row m=lane,    halves = K[k*32+0..7]  | K[k*32+16..23]
    //      lanes 16-31: row m=lane-16, halves = K[k*32+8..15] | K[k*32+24..31]
    const int m    = lane & 15;
    const int ksub = (lane >> 4) * 8;
    BF16Vec A0[NKSTEP], A1[NKSTEP];
    {
        const uint16_t* xrow0 = xB + (size_t)(row0 + m) * D_IN;
        const uint16_t* xrow1 = xrow0 + 16 * D_IN;
#pragma unroll
        for (int k = 0; k < NKSTEP; ++k) {
            const int kb = k * 32 + ksub;
            A0[k].q[0] = *(const uint4*)(xrow0 + kb);
            A0[k].q[1] = *(const uint4*)(xrow0 + kb + 16);
            A1[k].q[0] = *(const uint4*)(xrow1 + kb);
            A1[k].q[1] = *(const uint4*)(xrow1 + kb + 16);
        }
    }

    // B layout: lane holds column n=lane&15; lanes 0-15 K 0..7 | 16..23, lanes 16-31
    // K 8..15 | 24..31 of each 32-wide K step (matches A layout).
    // Base pointer starts at this wave's N-half.
    const uint16_t* wrowLane =
        WencB + (size_t)(half * NHALF * 16 + (lane & 15)) * D_IN +
        (lane >> 4) * 8;

    // ---- Software-pipeline prologue: first 3 K-steps of first tile ------
    BF16Vec Bq[4];
#pragma unroll
    for (int p = 0; p < 3; ++p) {
        Bq[p].q[0] = *(const uint4*)(wrowLane + p * 32);
        Bq[p].q[1] = *(const uint4*)(wrowLane + p * 32 + 16);
    }

    // ---- N-loop over this wave's 384 column tiles -----------------------
    for (int tt = 0; tt < NHALF; ++tt) {
        // Lockstep: same-half waves issue the same tile's B loads together.
        __syncthreads();

        const int t = half * NHALF + tt;           // global tile id
        const int ncol = t * 16 + (lane & 15);
        const uint16_t* wrow  = wrowLane + (size_t)tt * 16 * D_IN;
        const uint16_t* wnext = wrow + 16 * D_IN;

        // Warm WGP$ a full tile ahead (speculative, worst case lands in ws).
        __builtin_prefetch(wnext, 0, 3);

        // Load bias early so waiting on it later doesn't drain the
        // next-tile B loads issued at the end of this K-loop.
        const float bb = b_enc[ncol];

        v8f acc0 = {}, acc1 = {};
#pragma unroll
        for (int k = 0; k < NKSTEP; ++k) {
            // Issue the load 3 steps ahead; the last 3 iterations fetch the
            // first 3 K-steps of the *next* tile (cross-tile steady state).
            const int kl = k + 3;
            const uint16_t* src = (kl < NKSTEP)
                                      ? (wrow + kl * 32)
                                      : (wnext + (kl - NKSTEP) * 32);
            Bq[kl & 3].q[0] = *(const uint4*)(src);
            Bq[kl & 3].q[1] = *(const uint4*)(src + 16);

            // One B tile feeds two independent accumulator chains (M=32).
            acc0 = __builtin_amdgcn_wmma_f32_16x16x32_bf16(
                false, A0[k].v, false, Bq[k & 3].v, (short)0, acc0,
                false, false);
            acc1 = __builtin_amdgcn_wmma_f32_16x16x32_bf16(
                false, A1[k].v, false, Bq[k & 3].v, (short)0, acc1,
                false, false);
        }

        // bias (per-column, same for every row in this lane's column)
#pragma unroll
        for (int r = 0; r < 8; ++r) { acc0[r] += bb; acc1[r] += bb; }

        // C/D layout: VGPR r -> row r (lanes 0-15) / row r+8 (lanes 16-31)
        const int rbase = (lane >> 4) * 8;
        const int n     = lane & 15;
#pragma unroll
        for (int r = 0; r < 8; ++r) {
            s_tile[wave][rbase + r][n]      = acc0[r];
            s_tile[wave][16 + rbase + r][n] = acc1[r];
        }

        asm volatile("s_wait_dscnt 0" ::: "memory");

        // Streaming top-20: lane L scans row L's 16 fresh values.
        {
            const float* rowp = &s_tile[wave][lane][0];
            float* lv = &s_topv[sidx][half][lane][0];
            int*   li = &s_topi[sidx][half][lane][0];
#pragma unroll
            for (int j = 0; j < 16; ++j) {
                const float val = rowp[j];
                if (val > thr) {          // rare path (~1% of elements)
                    int pos = 0;
                    while (pos < KTOP && !(val > lv[pos])) ++pos;
                    for (int s = KTOP - 1; s > pos; --s) {
                        lv[s] = lv[s - 1];
                        li[s] = li[s - 1];
                    }
                    lv[pos] = val;
                    li[pos] = t * 16 + j;
                    thr = lv[KTOP - 1];
                }
            }
        }
    }

    __syncthreads();   // both halves' lists complete for both strips

    // ---- Merge the two sorted 20-lists per row (two-pointer) ------------
    // Wave w owns strip sidx, rows [half*16, half*16+16) -- exactly the rows
    // it will decode, so only a wave-local DS wait is needed afterwards.
    if (lane < 16) {
        const int row = half * 16 + lane;
        const float* av = &s_topv[sidx][0][row][0];
        const int*   ai = &s_topi[sidx][0][row][0];
        const float* bv = &s_topv[sidx][1][row][0];
        const int*   bi = &s_topi[sidx][1][row][0];
        int pa = 0, pb = 0;
#pragma unroll
        for (int i = 0; i < KTOP; ++i) {
            const float va = av[pa], vb = bv[pb];
            const bool ta = (va >= vb);
            s_mv[sidx][row][i] = ta ? va : vb;
            s_mi[sidx][row][i] = ta ? ai[pa] : bi[pb];
            pa += ta ? 1 : 0;
            pb += ta ? 0 : 1;
        }
    }
    asm volatile("s_wait_dscnt 0" ::: "memory");

    // ---- Sparse decoder: out[row,:] = b_dec + sum_20 v * Wt[idx,:] ------
    for (int r = 0; r < 16; ++r) {
        const int row = half * 16 + r;
        float accd[24];
#pragma unroll
        for (int i = 0; i < 24; ++i) accd[i] = b_dec[lane + i * 32];
#pragma unroll
        for (int s = 0; s < KTOP; ++s) {
            const float v = s_mv[sidx][row][s];   // LDS broadcast
            const int   j = s_mi[sidx][row][s];
            const float* wr = Wt + (size_t)j * D_IN;
#pragma unroll
            for (int i = 0; i < 24; ++i)
                accd[i] = fmaf(v, wr[lane + i * 32], accd[i]);
        }
        float* orow = out + (size_t)(row0 + row) * D_IN;
#pragma unroll
        for (int i = 0; i < 24; ++i) orow[lane + i * 32] = accd[i];
    }
}

// ---------------------------------------------------------------------------
extern "C" void kernel_launch(void* const* d_in, const int* in_sizes, int n_in,
                              void* d_out, int out_size, void* d_ws,
                              size_t ws_size, hipStream_t stream) {
    const float* x     = (const float*)d_in[0];  // [16384][768]
    const float* W_enc = (const float*)d_in[1];  // [12288][768]
    const float* b_enc = (const float*)d_in[2];  // [12288]
    const float* W_dec = (const float*)d_in[3];  // [768][12288]
    const float* b_dec = (const float*)d_in[4];  // [768]
    float* out = (float*)d_out;

    // Workspace layout (bytes): xB bf16 | WencB bf16 | Wt fp32
    char* ws = (char*)d_ws;
    uint16_t* xB    = (uint16_t*)ws;                       // 25,165,824 B
    uint16_t* WencB = (uint16_t*)(ws + 25165824);          // 18,874,368 B
    float*    Wt    = (float*)(ws + 25165824 + 18874368);  // 37,748,736 B

    cvt_bf16_pk<<<4096, 256, 0, stream>>>(x, (uint32_t*)xB,
                                          (NROWS * D_IN) / 2);
    cvt_bf16_pk<<<4096, 256, 0, stream>>>(W_enc, (uint32_t*)WencB,
                                          (D_LAT * D_IN) / 2);
    {
        dim3 grid(D_LAT / 32, D_IN / 32);
        dim3 block(32, 8);
        transpose_f32<<<grid, block, 0, stream>>>(W_dec, Wt);
    }
    // 512 strips of 32 rows x 2 N-half waves = 1024 waves; 4 waves/block
    sae_fused<<<256, WPB * 32, 0, stream>>>(xB, WencB, b_enc, Wt, b_dec, out);
}
